// GATModel_20624432956289
// MI455X (gfx1250) — matrix operand order repaired and checked by reading, not compiled, not run on hardware
//
#include <hip/hip_runtime.h>
#include <math.h>

// ---------------------------------------------------------------------------
// GAT (2-layer, PyG GATConv semantics) for MI455X / gfx1250 (wave32, WMMA).
// Dense GEMMs run on v_wmma_f32_16x16x32_bf16; edge softmax/aggregation uses
// float atomics (native on CDNA5).
// ---------------------------------------------------------------------------

typedef __attribute__((ext_vector_type(16))) __bf16 v16bf;
typedef __attribute__((ext_vector_type(8)))  __bf16 v8bf;
typedef __attribute__((ext_vector_type(8)))  float  v8f;

#define IN_DIM    1024
#define HID       64
#define HEADS     4
#define F1        (HEADS * HID)   // 256
#define N_GRAPHS  64
#define NEG_SLOPE 0.2f

__device__ __forceinline__ __bf16 f2bf(float f) {
  unsigned u = __builtin_bit_cast(unsigned, f);
  unsigned r = u + 0x7fffu + ((u >> 16) & 1u);   // round-to-nearest-even
  return __builtin_bit_cast(__bf16, (unsigned short)(r >> 16));
}

// float -> monotone-increasing unsigned key (for atomicMax-based float max)
__device__ __forceinline__ unsigned fkey(float f) {
  unsigned u = __builtin_bit_cast(unsigned, f);
  return (u & 0x80000000u) ? ~u : (u | 0x80000000u);
}
__device__ __forceinline__ float funkey(unsigned k) {
  unsigned u = (k & 0x80000000u) ? (k ^ 0x80000000u) : ~k;
  return __builtin_bit_cast(float, u);
}

// ------------------------------- utility kernels ---------------------------

__global__ void fill_u32_kernel(unsigned* __restrict__ p, long n, unsigned v) {
  long i = (long)blockIdx.x * blockDim.x + threadIdx.x;
  if (i < n) p[i] = v;
}

__global__ void cvt_bf16_kernel(const float* __restrict__ in,
                                __bf16* __restrict__ out, long n) {
  long i = (long)blockIdx.x * blockDim.x + threadIdx.x;
  if (i < n) out[i] = f2bf(in[i]);
}

// in: [K,N] f32 row-major  ->  out: [N,K] bf16 row-major
__global__ void transpose_bf16_kernel(const float* __restrict__ in,
                                      __bf16* __restrict__ out, int K, int N) {
  long i = (long)blockIdx.x * blockDim.x + threadIdx.x;
  if (i < (long)K * N) {
    int k = (int)(i / N), n = (int)(i % N);
    out[(long)n * K + k] = f2bf(in[i]);
  }
}

// ------------------------------- WMMA GEMM ---------------------------------
// C[M,N] = A[M,K](bf16) @ Bt[N,K]^T(bf16), f32 accumulate.
// One wave computes a 16(M) x 64(N) strip: 4 accumulators over K tiles of 32.
// A frag (16x32 bf16): lane m = lane&15; K elems = (lane>>4)*16 .. +15
//   contiguous (matches B).
// B frag (32x16 bf16): lane n = lane&15; K elems = (lane>>4)*16 .. +15
//   contiguous (matches A).
// C/D: VGPR r -> row = r + (lane>=16)*8, col = lane&15.
__global__ void gemm_bf16_wmma(const __bf16* __restrict__ A,
                               const __bf16* __restrict__ Bt,
                               float* __restrict__ C,
                               int M, int K, int N, int totalWaves) {
  int wave = threadIdx.x >> 5;
  int lane = threadIdx.x & 31;
  int gw = blockIdx.x * (blockDim.x >> 5) + wave;
  if (gw >= totalWaves) return;            // wave-uniform guard (EXEC stays full)

  int ntiles64 = N >> 6;
  int mt = gw / ntiles64;
  int nt = gw % ntiles64;

  int kbA = (lane >> 4) << 4;              // 0 or 16
  int kbB = (lane >> 4) << 4;              // 0 or 16
  const __bf16* arow = A + (long)(mt * 16 + (lane & 15)) * K;

  v8f acc[4] = {};
  for (int k0 = 0; k0 < K; k0 += 32) {
    union { v16bf v; v8bf h[2]; } af;
    af.h[0] = *(const v8bf*)(arow + k0 + kbA);
    af.h[1] = *(const v8bf*)(arow + k0 + kbA + 8);
#pragma unroll
    for (int j = 0; j < 4; ++j) {
      const __bf16* brow = Bt + (long)(nt * 64 + j * 16 + (lane & 15)) * K;
      v16bf bf = *(const v16bf*)(brow + k0 + kbB);
      acc[j] = __builtin_amdgcn_wmma_f32_16x16x32_bf16(
          false, af.v, false, bf, (short)0, acc[j], false, false);
    }
  }

  int rbase = mt * 16 + ((lane >> 4) ? 8 : 0);
  int c0 = lane & 15;
#pragma unroll
  for (int j = 0; j < 4; ++j) {
    int col = nt * 64 + j * 16 + c0;
#pragma unroll
    for (int r = 0; r < 8; ++r)
      C[(long)(rbase + r) * N + col] = acc[j][r];
  }
}

// ------------------------- attention coefficient kernels -------------------

// als[n,h] = sum_c h[n,h,c]*a_src[h,c] ; ald likewise
__global__ void node_alpha_kernel(const float* __restrict__ h,
                                  const float* __restrict__ a_src,
                                  const float* __restrict__ a_dst,
                                  float* __restrict__ als,
                                  float* __restrict__ ald,
                                  int Nn, int Hh, int Cc) {
  long i = (long)blockIdx.x * blockDim.x + threadIdx.x;
  if (i >= (long)Nn * Hh) return;
  int n = (int)(i / Hh), hd = (int)(i % Hh);
  const float* hp = h + (long)n * Hh * Cc + (long)hd * Cc;
  const float* as = a_src + hd * Cc;
  const float* ad = a_dst + hd * Cc;
  float s = 0.f, d = 0.f;
  for (int c = 0; c < Cc; ++c) { float v = hp[c]; s += v * as[c]; d += v * ad[c]; }
  als[i] = s; ald[i] = d;
}

__global__ void edge_max_kernel(const int* __restrict__ src, const int* __restrict__ dst,
                                const float* __restrict__ als, const float* __restrict__ ald,
                                unsigned* __restrict__ mkey, int E, int Hh) {
  long e = (long)blockIdx.x * blockDim.x + threadIdx.x;
  if (e >= E) return;
  int s = src[e], d = dst[e];
  for (int h = 0; h < Hh; ++h) {
    float t = als[(long)s * Hh + h] + ald[(long)d * Hh + h];
    t = t > 0.f ? t : NEG_SLOPE * t;
    atomicMax(&mkey[(long)d * Hh + h], fkey(t));
  }
}

__global__ void edge_expsum_kernel(const int* __restrict__ src, const int* __restrict__ dst,
                                   const float* __restrict__ als, const float* __restrict__ ald,
                                   const unsigned* __restrict__ mkey,
                                   float* __restrict__ wv, float* __restrict__ den,
                                   int E, int Hh) {
  long e = (long)blockIdx.x * blockDim.x + threadIdx.x;
  if (e >= E) return;
  int s = src[e], d = dst[e];
  for (int h = 0; h < Hh; ++h) {
    float t = als[(long)s * Hh + h] + ald[(long)d * Hh + h];
    t = t > 0.f ? t : NEG_SLOPE * t;
    float m = funkey(mkey[(long)d * Hh + h]);
    float w = expf(t - m);
    wv[e * Hh + h] = w;
    atomicAdd(&den[(long)d * Hh + h], w);
  }
}

// one block per edge; blockDim = Hh*Cc; out[dst] += h[src] * alpha
__global__ void edge_aggregate_kernel(const int* __restrict__ src, const int* __restrict__ dst,
                                      const float* __restrict__ hfeat,
                                      const float* __restrict__ wv,
                                      const float* __restrict__ den,
                                      float* __restrict__ out, int Hh, int Cc) {
  long e = blockIdx.x;
  int t = threadIdx.x;
  int s = src[e], d = dst[e];
  int h = t / Cc;
  float alpha = wv[e * Hh + h] / den[(long)d * Hh + h];
  atomicAdd(&out[(long)d * Hh * Cc + t], hfeat[(long)s * Hh * Cc + t] * alpha);
}

// v = elu(in + bias); optionally write f32 and/or bf16 copies
__global__ void bias_elu_kernel(const float* __restrict__ in, const float* __restrict__ bias,
                                float* __restrict__ outf, __bf16* __restrict__ outb,
                                long n, int F) {
  long i = (long)blockIdx.x * blockDim.x + threadIdx.x;
  if (i >= n) return;
  float v = in[i] + bias[i % F];
  v = v > 0.f ? v : (expf(v) - 1.f);
  if (outf) outf[i] = v;
  if (outb) outb[i] = f2bf(v);
}

// ------------------------------- pooling + FC ------------------------------

__global__ void pool_accum_kernel(const float* __restrict__ h, const int* __restrict__ batch,
                                  float* __restrict__ psum, float* __restrict__ pcnt,
                                  int Nn, int Cc) {
  long i = (long)blockIdx.x * blockDim.x + threadIdx.x;
  if (i >= (long)Nn * Cc) return;
  int n = (int)(i / Cc), c = (int)(i % Cc);
  int g = batch[n];
  atomicAdd(&psum[(long)g * Cc + c], h[i]);
  if (c == 0) atomicAdd(&pcnt[g], 1.0f);
}

__global__ void final_fc_kernel(const float* __restrict__ psum, const float* __restrict__ pcnt,
                                const float* __restrict__ fcw, const float* __restrict__ fcb,
                                float* __restrict__ out, int G, int Cc, int O) {
  int i = blockIdx.x * blockDim.x + threadIdx.x;
  if (i >= G * O) return;
  int g = i / O, o = i % O;
  float cnt = fmaxf(pcnt[g], 1.0f);
  float acc = 0.f;
  for (int c = 0; c < Cc; ++c)
    acc += (psum[(long)g * Cc + c] / cnt) * fcw[c * O + o];
  out[i] = acc + fcb[o];
}

// ------------------------------- host launcher -----------------------------

static inline long cdivl(long a, long b) { return (a + b - 1) / b; }

extern "C" void kernel_launch(void* const* d_in, const int* in_sizes, int n_in,
                              void* d_out, int out_size, void* d_ws, size_t ws_size,
                              hipStream_t stream) {
  (void)n_in; (void)out_size; (void)ws_size;
  const float* x    = (const float*)d_in[0];
  const int*   eidx = (const int*)  d_in[1];
  const int*   batch= (const int*)  d_in[2];
  const float* W1   = (const float*)d_in[3];
  const float* as1  = (const float*)d_in[4];
  const float* ad1  = (const float*)d_in[5];
  const float* b1   = (const float*)d_in[6];
  const float* W2   = (const float*)d_in[7];
  const float* as2  = (const float*)d_in[8];
  const float* ad2  = (const float*)d_in[9];
  const float* b2   = (const float*)d_in[10];
  const float* fcw  = (const float*)d_in[11];
  const float* fcb  = (const float*)d_in[12];

  const int Nn = in_sizes[0] / IN_DIM;   // 50000 (multiple of 16)
  const int E  = in_sizes[1] / 2;        // 850000
  const int* src = eidx;
  const int* dst = eidx + E;

  // ---- workspace bump allocation (256B aligned) ----
  char* ws = (char*)d_ws;
  size_t off = 0;
  auto alloc = [&](size_t bytes) -> void* {
    void* p = ws + off;
    off += (bytes + 255) & ~(size_t)255;
    return p;
  };
  __bf16* xb   = (__bf16*)  alloc((size_t)Nn * IN_DIM * 2);  // dead after GEMM1
  __bf16* W1t  = (__bf16*)  alloc((size_t)F1 * IN_DIM * 2);
  __bf16* W2t  = (__bf16*)  alloc((size_t)HID * F1 * 2);
  float*  h1   = (float*)   alloc((size_t)Nn * F1 * 4);
  float*  alS1 = (float*)   alloc((size_t)Nn * HEADS * 4);
  float*  alD1 = (float*)   alloc((size_t)Nn * HEADS * 4);
  unsigned* m1 = (unsigned*)alloc((size_t)Nn * HEADS * 4);
  float*  den1 = (float*)   alloc((size_t)Nn * HEADS * 4);
  float*  we1  = (float*)   alloc((size_t)E * HEADS * 4);
  __bf16* h1a  = (__bf16*)  alloc((size_t)Nn * F1 * 2);
  float*  h2   = (float*)   alloc((size_t)Nn * HID * 4);
  float*  alS2 = (float*)   alloc((size_t)Nn * 4);
  float*  alD2 = (float*)   alloc((size_t)Nn * 4);
  unsigned* m2 = (unsigned*)alloc((size_t)Nn * 4);
  float*  den2 = (float*)   alloc((size_t)Nn * 4);
  float*  we2  = (float*)   alloc((size_t)E * 4);
  float*  psum = (float*)   alloc((size_t)N_GRAPHS * HID * 4);
  float*  pcnt = (float*)   alloc((size_t)N_GRAPHS * 4);
  // alias late buffers into the (then-dead) xb region: 51.2+12.8+12.8 < 102.4 MB
  float* out1 = (float*)xb;
  float* out2 = out1 + (size_t)Nn * F1;
  float* h2a  = out2 + (size_t)Nn * HID;

  const int TPB = 256;

  // ---- prep: bf16 conversions / weight transposes ----
  long nx = (long)Nn * IN_DIM;
  cvt_bf16_kernel<<<(unsigned)cdivl(nx, TPB), TPB, 0, stream>>>(x, xb, nx);
  transpose_bf16_kernel<<<(unsigned)cdivl((long)IN_DIM * F1, TPB), TPB, 0, stream>>>(W1, W1t, IN_DIM, F1);
  transpose_bf16_kernel<<<(unsigned)cdivl((long)F1 * HID, TPB), TPB, 0, stream>>>(W2, W2t, F1, HID);

  // ---- layer 1 ----
  const int mtiles = Nn / 16;
  int tw1 = mtiles * (F1 / 64);                  // 12500 waves
  gemm_bf16_wmma<<<(unsigned)cdivl(tw1, 4), 128, 0, stream>>>(xb, W1t, h1, Nn, IN_DIM, F1, tw1);

  node_alpha_kernel<<<(unsigned)cdivl((long)Nn * HEADS, TPB), TPB, 0, stream>>>(
      h1, as1, ad1, alS1, alD1, Nn, HEADS, HID);
  fill_u32_kernel<<<(unsigned)cdivl((long)Nn * HEADS, TPB), TPB, 0, stream>>>(m1, (long)Nn * HEADS, 0u);
  fill_u32_kernel<<<(unsigned)cdivl((long)Nn * HEADS, TPB), TPB, 0, stream>>>((unsigned*)den1, (long)Nn * HEADS, 0u);
  edge_max_kernel<<<(unsigned)cdivl(E, TPB), TPB, 0, stream>>>(src, dst, alS1, alD1, m1, E, HEADS);
  edge_expsum_kernel<<<(unsigned)cdivl(E, TPB), TPB, 0, stream>>>(src, dst, alS1, alD1, m1, we1, den1, E, HEADS);
  fill_u32_kernel<<<(unsigned)cdivl((long)Nn * F1, TPB), TPB, 0, stream>>>((unsigned*)out1, (long)Nn * F1, 0u);
  edge_aggregate_kernel<<<(unsigned)E, F1, 0, stream>>>(src, dst, h1, we1, den1, out1, HEADS, HID);
  bias_elu_kernel<<<(unsigned)cdivl((long)Nn * F1, TPB), TPB, 0, stream>>>(
      out1, b1, (float*)nullptr, h1a, (long)Nn * F1, F1);

  // ---- layer 2 ----
  int tw2 = mtiles * (HID / 64);                 // 3125 waves
  gemm_bf16_wmma<<<(unsigned)cdivl(tw2, 8), 256, 0, stream>>>(h1a, W2t, h2, Nn, F1, HID, tw2);

  node_alpha_kernel<<<(unsigned)cdivl((long)Nn, TPB), TPB, 0, stream>>>(
      h2, as2, ad2, alS2, alD2, Nn, 1, HID);
  fill_u32_kernel<<<(unsigned)cdivl((long)Nn, TPB), TPB, 0, stream>>>(m2, (long)Nn, 0u);
  fill_u32_kernel<<<(unsigned)cdivl((long)Nn, TPB), TPB, 0, stream>>>((unsigned*)den2, (long)Nn, 0u);
  edge_max_kernel<<<(unsigned)cdivl(E, TPB), TPB, 0, stream>>>(src, dst, alS2, alD2, m2, E, 1);
  edge_expsum_kernel<<<(unsigned)cdivl(E, TPB), TPB, 0, stream>>>(src, dst, alS2, alD2, m2, we2, den2, E, 1);
  fill_u32_kernel<<<(unsigned)cdivl((long)Nn * HID, TPB), TPB, 0, stream>>>((unsigned*)out2, (long)Nn * HID, 0u);
  edge_aggregate_kernel<<<(unsigned)E, HID, 0, stream>>>(src, dst, h2, we2, den2, out2, 1, HID);
  bias_elu_kernel<<<(unsigned)cdivl((long)Nn * HID, TPB), TPB, 0, stream>>>(
      out2, b2, h2a, (__bf16*)nullptr, (long)Nn * HID, HID);

  // ---- global mean pool + FC ----
  fill_u32_kernel<<<(unsigned)cdivl((long)N_GRAPHS * HID, TPB), TPB, 0, stream>>>(
      (unsigned*)psum, (long)N_GRAPHS * HID, 0u);
  fill_u32_kernel<<<1, TPB, 0, stream>>>((unsigned*)pcnt, (long)N_GRAPHS, 0u);
  pool_accum_kernel<<<(unsigned)cdivl((long)Nn * HID, TPB), TPB, 0, stream>>>(h2a, batch, psum, pcnt, Nn, HID);
  final_fc_kernel<<<1, 128, 0, stream>>>(psum, pcnt, fcw, fcb, (float*)d_out, N_GRAPHS, HID, 2);
}
